// Decoder_16767552323794
// MI455X (gfx1250) — matrix operand order repaired and checked by reading, not compiled, not run on hardware
//
#include <hip/hip_runtime.h>

// ---------------------------------------------------------------------------
// LAS attention decoder for gfx1250 (MI455X).
//   V=8000 H=512 KS=VS=128 T=500 N=64 L=200
// Strategy:
//  - Weights packed once per call into bf16 WMMA-B fragment order.
//  - Activations produced in f32, stored bf16 into concatenated "A" buffers.
//  - GEMMs via v_wmma_f32_16x16x32_bf16, f32 accumulate; one wave = 64x16 strip.
//  - Gates / softmax / attention context in f32 VALU (latency-bound, L2-resident).
// ---------------------------------------------------------------------------

#define NB   64     // batch
#define HID  512
#define KSZ  128
#define VSZ  128
#define TT   500
#define LL   200
#define VOC  8000

#define KA1  1152   // [embed 512 | ctx 128 | h1 512]
#define KA2  640    // [h1 512 | h2 128]
#define KA3  256    // [h2 128 | ctx 128]
#define G1   2048   // 4*H
#define G2   512    // 4*KS

typedef __attribute__((ext_vector_type(16))) __bf16 v16bf;
typedef __attribute__((ext_vector_type(8)))  __bf16 v8bf;
typedef __attribute__((ext_vector_type(8)))  float  v8f;

__device__ __forceinline__ v16bf cat8(v8bf lo, v8bf hi) {
    return __builtin_shufflevector(lo, hi, 0,1,2,3,4,5,6,7,8,9,10,11,12,13,14,15);
}
__device__ __forceinline__ float sigf(float x) { return 1.0f / (1.0f + __expf(-x)); }

// ---------------------------------------------------------------------------
// Weight pack: W row-major (Nout x Kin), logically concat of src0 (Nout x K0)
// and src1 (Nout x (Kin-K0)) along K.  Output layout: [ntile][ktile][lane][16]
// bf16 so each lane's 16-element B fragment is one contiguous 32-byte load.
// Fragment mapping (16-bit B 32x16): N=lane%16, h=lane/16; element e<8 -> K=kt*32+8h+e,
// e>=8 -> K=kt*32+16+8h+(e-8).
// ---------------------------------------------------------------------------
__global__ void pack_weights(const float* __restrict__ src0,
                             const float* __restrict__ src1,
                             int K0, int Kin, long total,
                             __bf16* __restrict__ pb) {
    long idx = (long)blockIdx.x * blockDim.x + threadIdx.x;
    if (idx >= total) return;
    int KT   = Kin >> 5;
    int e    = (int)(idx & 15);
    int lane = (int)((idx >> 4) & 31);
    long rest = idx >> 9;
    int kt   = (int)(rest % KT);
    int nt   = (int)(rest / KT);
    int nrow = nt * 16 + (lane & 15);
    int k    = kt * 32 + (lane >> 4) * 8 + (e & 7) + (e >> 3) * 16;
    float v  = (k < K0) ? src0[(long)nrow * K0 + k]
                        : src1[(long)nrow * (Kin - K0) + (k - K0)];
    pb[idx] = (__bf16)v;
}

__global__ void bias_sum(const float* __restrict__ a, const float* __restrict__ b,
                         float* __restrict__ o, int n) {
    int i = blockIdx.x * blockDim.x + threadIdx.x;
    if (i < n) o[i] = a[i] + b[i];
}

// init: A1 ctx slot = values[0], A1 h1 slot = 0, A2 = 0, c1 = 0, c2 = 0
__global__ void init_state(const float* __restrict__ values,
                           __bf16* __restrict__ A1, __bf16* __restrict__ A2,
                           float* __restrict__ c1, float* __restrict__ c2) {
    int idx = blockIdx.x * blockDim.x + threadIdx.x;     // 64*640
    if (idx >= NB * KA2) return;
    int n = idx / KA2, j = idx % KA2;                    // j in [0,640)
    int ja = j + HID;                                    // A1 slot [512,1152)
    float v = (ja < HID + VSZ) ? values[n * VSZ + (ja - HID)] : 0.0f;
    A1[n * KA1 + ja] = (__bf16)v;
    A2[idx] = (__bf16)0.0f;
    if (idx < NB * HID) c1[idx] = 0.0f;
    if (idx < NB * KSZ) c2[idx] = 0.0f;
}

// embedding gather for step t -> A1[:, 0:512] (bf16)
__global__ void embed_step(const int* __restrict__ text, const float* __restrict__ emb,
                           __bf16* __restrict__ A1, int t) {
    int idx = blockIdx.x * blockDim.x + threadIdx.x;     // 64*512
    if (idx >= NB * HID) return;
    int n = idx >> 9, h = idx & (HID - 1);
    int tok = text[n * LL + t];
    float v = (tok == 0) ? 0.0f : emb[(long)tok * HID + h];
    A1[n * KA1 + h] = (__bf16)v;
}

// ---------------------------------------------------------------------------
// WMMA GEMM: Z(64 x Nout) = A(64 x K, bf16 row-major) x Wpacked.
// One wave per 16-wide N strip over all 64 rows (4 accumulators reuse B frag).
// Grid must supply exactly Nout/16 waves (blockDim=128 -> 4 waves/block).
// ---------------------------------------------------------------------------
__global__ __launch_bounds__(128) void wmma_gemm64(
        const __bf16* __restrict__ A, const __bf16* __restrict__ PB,
        float* __restrict__ Z, int K, int Nout) {
    int wv   = blockIdx.x * 4 + (threadIdx.x >> 5);
    int lane = threadIdx.x & 31;
    if (wv * 16 >= Nout) return;
    int KT = K >> 5;
    // A fragment base: row = lane%16 (+16*mt), K half-select by lane/16
    const __bf16* a0 = A + (lane & 15) * K + (lane >> 4) * 8;
    const v16bf*  b0 = (const v16bf*)PB + (size_t)wv * KT * 32 + lane;
    v8f acc0 = {}, acc1 = {}, acc2 = {}, acc3 = {};
    for (int kk = 0; kk < KT; ++kk) {
        v16bf bf = b0[kk * 32];
        if (kk + 1 < KT) __builtin_prefetch(b0 + (kk + 1) * 32, 0, 1);
        const __bf16* ap = a0 + kk * 32;
        v16bf af0 = cat8(*(const v8bf*)(ap),             *(const v8bf*)(ap + 16));
        v16bf af1 = cat8(*(const v8bf*)(ap + 16 * K),    *(const v8bf*)(ap + 16 * K + 16));
        v16bf af2 = cat8(*(const v8bf*)(ap + 32 * K),    *(const v8bf*)(ap + 32 * K + 16));
        v16bf af3 = cat8(*(const v8bf*)(ap + 48 * K),    *(const v8bf*)(ap + 48 * K + 16));
        acc0 = __builtin_amdgcn_wmma_f32_16x16x32_bf16(false, af0, false, bf, (short)0, acc0, false, false);
        acc1 = __builtin_amdgcn_wmma_f32_16x16x32_bf16(false, af1, false, bf, (short)0, acc1, false, false);
        acc2 = __builtin_amdgcn_wmma_f32_16x16x32_bf16(false, af2, false, bf, (short)0, acc2, false, false);
        acc3 = __builtin_amdgcn_wmma_f32_16x16x32_bf16(false, af3, false, bf, (short)0, acc3, false, false);
    }
    // D layout: lane<16 -> (M=r, N=lane); lane>=16 -> (M=8+r, N=lane-16)
    int col  = wv * 16 + (lane & 15);
    int roff = (lane >> 4) * 8;
    float* z = Z + col;
#pragma unroll
    for (int r = 0; r < 8; ++r) {
        int row = r + roff;
        z[(size_t)(row +  0) * Nout] = acc0[r];
        z[(size_t)(row + 16) * Nout] = acc1[r];
        z[(size_t)(row + 32) * Nout] = acc2[r];
        z[(size_t)(row + 48) * Nout] = acc3[r];
    }
}

// Output projection: pred(64 x 8000) = A3(64x256) x W_out^T + b_out -> out[n][t][v]
__global__ __launch_bounds__(128) void wmma_out(
        const __bf16* __restrict__ A, const __bf16* __restrict__ PB,
        const float* __restrict__ bout, float* __restrict__ out, int t) {
    const int K = KA3, KT = KA3 / 32;
    int wv   = blockIdx.x * 4 + (threadIdx.x >> 5);
    int lane = threadIdx.x & 31;
    if (wv * 16 >= VOC) return;
    const __bf16* a0 = A + (lane & 15) * K + (lane >> 4) * 8;
    const v16bf*  b0 = (const v16bf*)PB + (size_t)wv * KT * 32 + lane;
    v8f acc0 = {}, acc1 = {}, acc2 = {}, acc3 = {};
#pragma unroll
    for (int kk = 0; kk < KT; ++kk) {
        v16bf bf = b0[kk * 32];
        const __bf16* ap = a0 + kk * 32;
        v16bf af0 = cat8(*(const v8bf*)(ap),          *(const v8bf*)(ap + 16));
        v16bf af1 = cat8(*(const v8bf*)(ap + 16 * K), *(const v8bf*)(ap + 16 * K + 16));
        v16bf af2 = cat8(*(const v8bf*)(ap + 32 * K), *(const v8bf*)(ap + 32 * K + 16));
        v16bf af3 = cat8(*(const v8bf*)(ap + 48 * K), *(const v8bf*)(ap + 48 * K + 16));
        acc0 = __builtin_amdgcn_wmma_f32_16x16x32_bf16(false, af0, false, bf, (short)0, acc0, false, false);
        acc1 = __builtin_amdgcn_wmma_f32_16x16x32_bf16(false, af1, false, bf, (short)0, acc1, false, false);
        acc2 = __builtin_amdgcn_wmma_f32_16x16x32_bf16(false, af2, false, bf, (short)0, acc2, false, false);
        acc3 = __builtin_amdgcn_wmma_f32_16x16x32_bf16(false, af3, false, bf, (short)0, acc3, false, false);
    }
    int col  = wv * 16 + (lane & 15);
    int roff = (lane >> 4) * 8;
    float bias = bout[col];
    float* o = out + (size_t)t * VOC + col;
#pragma unroll
    for (int r = 0; r < 8; ++r) {
        int row = r + roff;
        o[(size_t)(row +  0) * LL * VOC] = acc0[r] + bias;
        o[(size_t)(row + 16) * LL * VOC] = acc1[r] + bias;
        o[(size_t)(row + 32) * LL * VOC] = acc2[r] + bias;
        o[(size_t)(row + 48) * LL * VOC] = acc3[r] + bias;
    }
}

// LSTM1 gates (PyTorch order i,f,g,o): consumes z1(64x2048), updates c1,
// writes h1 bf16 into A1[:,640:1152] and A2[:,0:512].
__global__ void gate1(const float* __restrict__ z1, const float* __restrict__ bias,
                      float* __restrict__ c1, __bf16* __restrict__ A1,
                      __bf16* __restrict__ A2) {
    int idx = blockIdx.x * blockDim.x + threadIdx.x;     // 64*512
    if (idx >= NB * HID) return;
    int n = idx >> 9, j = idx & (HID - 1);
    const float* z = z1 + (size_t)n * G1;
    float gi = z[j]            + bias[j];
    float gf = z[HID + j]      + bias[HID + j];
    float gg = z[2 * HID + j]  + bias[2 * HID + j];
    float go = z[3 * HID + j]  + bias[3 * HID + j];
    float c  = sigf(gf) * c1[idx] + sigf(gi) * tanhf(gg);
    float h  = sigf(go) * tanhf(c);
    c1[idx] = c;
    A1[n * KA1 + (HID + VSZ) + j] = (__bf16)h;
    A2[n * KA2 + j]               = (__bf16)h;
}

// LSTM2 gates: consumes z2(64x512), updates c2, writes h2 into A2[:,512:640],
// A3[:,0:128] (bf16) and h2f (f32 for attention).
__global__ void gate2(const float* __restrict__ z2, const float* __restrict__ bias,
                      float* __restrict__ c2, __bf16* __restrict__ A2,
                      __bf16* __restrict__ A3, float* __restrict__ h2f) {
    int idx = blockIdx.x * blockDim.x + threadIdx.x;     // 64*128
    if (idx >= NB * KSZ) return;
    int n = idx >> 7, j = idx & (KSZ - 1);
    const float* z = z2 + (size_t)n * G2;
    float gi = z[j]            + bias[j];
    float gf = z[KSZ + j]      + bias[KSZ + j];
    float gg = z[2 * KSZ + j]  + bias[2 * KSZ + j];
    float go = z[3 * KSZ + j]  + bias[3 * KSZ + j];
    float c  = sigf(gf) * c2[idx] + sigf(gi) * tanhf(gg);
    float h  = sigf(go) * tanhf(c);
    c2[idx] = c;
    A2[n * KA2 + HID + j] = (__bf16)h;
    A3[n * KA3 + j]       = (__bf16)h;
    h2f[idx] = h;
}

// Attention: one block per batch row. energy -> masked softmax -> context.
// key (T,N,KS), values (T,N,VS), f32; writes ctx bf16 to A1[:,512:640], A3[:,128:256].
__global__ __launch_bounds__(256) void attn_step(
        const float* __restrict__ key, const float* __restrict__ values,
        const int* __restrict__ lens, const float* __restrict__ h2f,
        __bf16* __restrict__ A1, __bf16* __restrict__ A3) {
    __shared__ float e[TT];
    __shared__ float red[256];
    int n = blockIdx.x, tid = threadIdx.x;
    int len = lens[n];
    const float4* hv = (const float4*)(h2f + n * KSZ);
    for (int t = tid; t < TT; t += 256) {
        const float4* kv = (const float4*)(key + ((size_t)t * NB + n) * KSZ);
        float s = 0.0f;
#pragma unroll
        for (int q = 0; q < KSZ / 4; ++q) {
            float4 a = kv[q], b = hv[q];
            s += a.x * b.x + a.y * b.y + a.z * b.z + a.w * b.w;
        }
        e[t] = (t < len) ? s : -1e9f;
    }
    __syncthreads();
    float m = -1e30f;
    for (int t = tid; t < TT; t += 256) m = fmaxf(m, e[t]);
    red[tid] = m; __syncthreads();
    for (int s = 128; s > 0; s >>= 1) {
        if (tid < s) red[tid] = fmaxf(red[tid], red[tid + s]);
        __syncthreads();
    }
    m = red[0]; __syncthreads();
    float sum = 0.0f;
    for (int t = tid; t < TT; t += 256) { float x = __expf(e[t] - m); e[t] = x; sum += x; }
    red[tid] = sum; __syncthreads();
    for (int s = 128; s > 0; s >>= 1) {
        if (tid < s) red[tid] += red[tid + s];
        __syncthreads();
    }
    float inv = 1.0f / red[0]; __syncthreads();
    for (int t = tid; t < TT; t += 256) e[t] *= inv;
    __syncthreads();
    int v = tid & (VSZ - 1), g = tid >> 7;               // 2 groups split T
    float s = 0.0f;
    for (int t = g; t < TT; t += 2)
        s += e[t] * values[((size_t)t * NB + n) * VSZ + v];
    red[tid] = s; __syncthreads();
    if (tid < VSZ) {
        float c = red[tid] + red[tid + VSZ];
        A1[n * KA1 + HID + v] = (__bf16)c;
        A3[n * KA3 + KSZ + v] = (__bf16)c;
    }
}

// ---------------------------------------------------------------------------
extern "C" void kernel_launch(void* const* d_in, const int* in_sizes, int n_in,
                              void* d_out, int out_size, void* d_ws, size_t ws_size,
                              hipStream_t stream) {
    const float* key    = (const float*)d_in[0];
    const float* values = (const float*)d_in[1];
    const int*   lens   = (const int*)  d_in[2];
    const int*   text   = (const int*)  d_in[3];
    const float* emb    = (const float*)d_in[4];
    const float* W_ih1  = (const float*)d_in[5];
    const float* W_hh1  = (const float*)d_in[6];
    const float* b_ih1  = (const float*)d_in[7];
    const float* b_hh1  = (const float*)d_in[8];
    const float* W_ih2  = (const float*)d_in[9];
    const float* W_hh2  = (const float*)d_in[10];
    const float* b_ih2  = (const float*)d_in[11];
    const float* b_hh2  = (const float*)d_in[12];
    const float* W_out  = (const float*)d_in[13];
    const float* b_out  = (const float*)d_in[14];
    float* out = (float*)d_out;

    size_t off = 0;
    auto alloc = [&](size_t bytes) -> void* {
        void* p = (char*)d_ws + off;
        off += (bytes + 255) & ~(size_t)255;
        return p;
    };
    __bf16* pb1   = (__bf16*)alloc((size_t)G1 * KA1 * 2);   // packed [W_ih1|W_hh1]
    __bf16* pb2   = (__bf16*)alloc((size_t)G2 * KA2 * 2);   // packed [W_ih2|W_hh2]
    __bf16* pbO   = (__bf16*)alloc((size_t)VOC * KA3 * 2);  // packed W_out
    float*  bias1 = (float*) alloc(G1 * 4);
    float*  bias2 = (float*) alloc(G2 * 4);
    __bf16* A1    = (__bf16*)alloc((size_t)NB * KA1 * 2);
    __bf16* A2    = (__bf16*)alloc((size_t)NB * KA2 * 2);
    __bf16* A3    = (__bf16*)alloc((size_t)NB * KA3 * 2);
    float*  z1    = (float*) alloc((size_t)NB * G1 * 4);
    float*  z2    = (float*) alloc((size_t)NB * G2 * 4);
    float*  c1    = (float*) alloc((size_t)NB * HID * 4);
    float*  c2    = (float*) alloc((size_t)NB * KSZ * 4);
    float*  h2f   = (float*) alloc((size_t)NB * KSZ * 4);
    (void)ws_size;

    // --- one-time (per call) weight packing + state init ---
    {
        long t1 = (long)G1 * KA1, t2 = (long)G2 * KA2, t3 = (long)VOC * KA3;
        pack_weights<<<dim3((t1 + 255) / 256), dim3(256), 0, stream>>>(W_ih1, W_hh1, HID + VSZ, KA1, t1, pb1);
        pack_weights<<<dim3((t2 + 255) / 256), dim3(256), 0, stream>>>(W_ih2, W_hh2, HID, KA2, t2, pb2);
        pack_weights<<<dim3((t3 + 255) / 256), dim3(256), 0, stream>>>(W_out, W_out, KA3, KA3, t3, pbO);
        bias_sum<<<dim3((G1 + 255) / 256), dim3(256), 0, stream>>>(b_ih1, b_hh1, bias1, G1);
        bias_sum<<<dim3((G2 + 255) / 256), dim3(256), 0, stream>>>(b_ih2, b_hh2, bias2, G2);
        init_state<<<dim3((NB * KA2 + 255) / 256), dim3(256), 0, stream>>>(values, A1, A2, c1, c2);
    }

    // --- sequential decode: 200 steps ---
    for (int t = 0; t < LL; ++t) {
        embed_step<<<dim3(NB * HID / 256), dim3(256), 0, stream>>>(text, emb, A1, t);
        // LSTM1: z1 = A1(64x1152) x pb1 -> (64x2048); 128 waves
        wmma_gemm64<<<dim3(G1 / 64), dim3(128), 0, stream>>>(A1, pb1, z1, KA1, G1);
        gate1<<<dim3(NB * HID / 256), dim3(256), 0, stream>>>(z1, bias1, c1, A1, A2);
        // LSTM2: z2 = A2(64x640) x pb2 -> (64x512); 32 waves
        wmma_gemm64<<<dim3(G2 / 64), dim3(128), 0, stream>>>(A2, pb2, z2, KA2, G2);
        gate2<<<dim3(NB * KSZ / 256), dim3(256), 0, stream>>>(z2, bias2, c2, A2, A3, h2f);
        attn_step<<<dim3(NB), dim3(256), 0, stream>>>(key, values, lens, h2f, A1, A3);
        // Output projection: 500 waves
        wmma_out<<<dim3(VOC / 64), dim3(128), 0, stream>>>(A3, pbO, b_out, out, t);
    }
}
